// WaveNet_31834297598099
// MI455X (gfx1250) — matrix operand order repaired and checked by reading, not compiled, not run on hardware
//
#include <hip/hip_runtime.h>
#include <math.h>

// ---- problem constants ----
#define QV 16293
#define RCH 24
#define DCH 32
#define SCH 128
#define NLAY 8
#define BB 4
#define LL 2048
#define TFULL 2047      // length of x0
#define OUTW 1792
#define XSTRIDE 32      // padded channel stride for x buffers
#define MROWS (BB*OUTW) // 7168
#define NTILES 1019     // ceil(16293/16)
#define MTILES (MROWS/16) // 448

typedef __attribute__((ext_vector_type(16))) _Float16 v16h;
typedef __attribute__((ext_vector_type(8)))  _Float16 v8h;
typedef __attribute__((ext_vector_type(8)))  float    v8f;

// ---------------- zero skip accumulator ----------------
__global__ void zero_kernel(float* __restrict__ p, int n) {
    int i = blockIdx.x * blockDim.x + threadIdx.x;
    if (i < n) p[i] = 0.0f;
}

// ---------------- embedding gather: x0[b,t,c] = preW[0][idx[b,t]][c] + preW[1][idx[b,t+1]][c]
__global__ void embed_kernel(const int* __restrict__ idx,
                             const float* __restrict__ preW,
                             float* __restrict__ x0) {
    int gid = blockIdx.x * blockDim.x + threadIdx.x;
    int total = BB * TFULL * RCH;
    if (gid >= total) return;
    int c  = gid % RCH;
    int bt = gid / RCH;
    int t  = bt % TFULL;
    int b  = bt / TFULL;
    int i0 = idx[b * LL + t];
    int i1 = idx[b * LL + t + 1];
    float v = preW[(size_t)i0 * RCH + c] + preW[(size_t)QV * RCH + (size_t)i1 * RCH + c];
    x0[((size_t)(b * TFULL + t)) * XSTRIDE + c] = v;
}

// ---------------- one WaveNet layer (weights staged in LDS, thread = one (b,t)) --------
__global__ __launch_bounds__(256) void layer_kernel(
    const float* __restrict__ x_in, float* __restrict__ x_out,
    float* __restrict__ skip_sum,
    const float* __restrict__ fw,   // [2][24][32]
    const float* __restrict__ gw,   // [2][24][32]
    const float* __restrict__ rw,   // [32][24]
    const float* __restrict__ sw,   // [32][128]
    int len_out, int skip_off)
{
    __shared__ float s_f[2 * RCH * DCH];   // 1536
    __shared__ float s_g[2 * RCH * DCH];   // 1536
    __shared__ float s_r[DCH * RCH];       // 768
    __shared__ float s_s[DCH * SCH];       // 4096  -> total 31744 B
    for (int j = threadIdx.x; j < 2 * RCH * DCH; j += blockDim.x) { s_f[j] = fw[j]; s_g[j] = gw[j]; }
    for (int j = threadIdx.x; j < DCH * RCH; j += blockDim.x) s_r[j] = rw[j];
    for (int j = threadIdx.x; j < DCH * SCH; j += blockDim.x) s_s[j] = sw[j];
    __syncthreads();

    int gid = blockIdx.x * blockDim.x + threadIdx.x;
    int total = BB * len_out;
    if (gid >= total) return;
    int b = gid / len_out;
    int t = gid % len_out;

    const float* xa = x_in + ((size_t)(b * TFULL + t)) * XSTRIDE;
    const float* xb = xa + XSTRIDE;
    float xav[RCH], xbv[RCH];
    #pragma unroll
    for (int c = 0; c < RCH; ++c) { xav[c] = xa[c]; xbv[c] = xb[c]; }

    float outv[DCH];
    #pragma unroll 4
    for (int d = 0; d < DCH; ++d) {
        float f = 0.f, g = 0.f;
        #pragma unroll
        for (int c = 0; c < RCH; ++c) {
            f += xav[c] * s_f[c * DCH + d] + xbv[c] * s_f[(RCH + c) * DCH + d];
            g += xav[c] * s_g[c * DCH + d] + xbv[c] * s_g[(RCH + c) * DCH + d];
        }
        outv[d] = tanhf(f) * (1.0f / (1.0f + expf(-g)));
    }

    // residual: x_out[t] = x_in[t+1] + out @ res_W
    float* xo = x_out + ((size_t)(b * TFULL + t)) * XSTRIDE;
    #pragma unroll 4
    for (int r = 0; r < RCH; ++r) {
        float acc = xbv[r];
        #pragma unroll
        for (int d = 0; d < DCH; ++d) acc += outv[d] * s_r[d * RCH + r];
        xo[r] = acc;
    }

    // skip: last OUTW positions accumulate out @ skip_W
    int p = t - skip_off;
    if (p >= 0 && p < OUTW) {
        float* sp = skip_sum + ((size_t)(b * OUTW + p)) * SCH;
        #pragma unroll 4
        for (int s = 0; s < SCH; ++s) {
            float acc = 0.f;
            #pragma unroll
            for (int d = 0; d < DCH; ++d) acc += outv[d] * s_s[d * SCH + s];
            sp[s] += acc;
        }
    }
}

// ---------------- h1 = relu(relu(skip) @ W1), store as f16 A-matrix (row-major, lda=128)
__global__ __launch_bounds__(128) void post_kernel(
    const float* __restrict__ skip, const float* __restrict__ w1,
    _Float16* __restrict__ Af16)
{
    __shared__ float h[SCH];
    int row = blockIdx.x;          // 0..MROWS-1
    int u   = threadIdx.x;         // 0..127
    float v = skip[(size_t)row * SCH + u];
    h[u] = v > 0.f ? v : 0.f;
    __syncthreads();
    float acc = 0.f;
    #pragma unroll 8
    for (int s = 0; s < SCH; ++s) acc += h[s] * w1[s * SCH + u];
    acc = acc > 0.f ? acc : 0.f;
    Af16[(size_t)row * SCH + u] = (_Float16)acc;
}

// ---------------- pack post_W2 (128 x Q, f32 row-major) into WMMA B-fragment layout ----
// B 32x16 f16 fragment: lane<16 -> K=kb*32+0..15, lane>=16 -> K=kb*32+16..31; N = lane&15.
// Per (ntile, kb, lane): 16 contiguous f16.
__global__ void packw2_kernel(const float* __restrict__ w2, _Float16* __restrict__ Bp) {
    int gid = blockIdx.x * blockDim.x + threadIdx.x;   // ((nt*4+kb)*32 + lane)
    if (gid >= NTILES * 4 * 32) return;
    int lane = gid & 31;
    int kb   = (gid >> 5) & 3;
    int nt   = gid >> 7;
    int n     = nt * 16 + (lane & 15);
    int kbase = kb * 32 + (lane >> 4) * 16;
    _Float16* dst = Bp + (size_t)gid * 16;
    #pragma unroll
    for (int e = 0; e < 16; ++e) {
        float v = (n < QV) ? w2[(size_t)(kbase + e) * QV + n] : 0.0f;
        dst[e] = (_Float16)v;
    }
}

// ---------------- final GEMM: out(7168 x 16293, f32) = A(7168x128 f16) @ W2(128x16293 f16)
// One wave per block. blockIdx.x = N-tile, blockIdx.y = chunk of 16 M-tiles.
// B fragments (K=128 -> 4 frags) loaded once, reused across 16 M-tiles.
// A fragments double-buffered so the 4 WMMAs issue back-to-back without
// per-step s_wait_loadcnt stalls, and next-tile loads overlap compute+stores.
__global__ __launch_bounds__(32) void gemm_kernel(
    const _Float16* __restrict__ A,
    const _Float16* __restrict__ Bp,
    float* __restrict__ out)
{
    int lane = threadIdx.x;        // 0..31
    int nt   = blockIdx.x;         // 0..NTILES-1
    int mblk = blockIdx.y;         // 0..27

    // B fragments: contiguous 32B per lane from packed layout
    v16h bf[4];
    const v16h* bsrc = ((const v16h*)Bp) + (size_t)nt * 4 * 32 + lane;
    #pragma unroll
    for (int kb = 0; kb < 4; ++kb) bf[kb] = bsrc[kb * 32];

    int ocol = nt * 16 + (lane & 15);
    int kgrp = lane >> 4;
    // A 16-bit 16x32 fragment layout: elems 0..15 -> K = 16*kgrp + e
    // (per 32-wide K block), matching the packed B layout.
    const _Float16* abase0 = A + ((size_t)mblk * 16 * 16 + (lane & 15)) * SCH + kgrp * 16;

    #define LOAD_A(dst, j)                                                    \
        {                                                                     \
            const _Float16* ab_ = abase0 + (size_t)(j) * 16 * SCH;            \
            _Pragma("unroll")                                                 \
            for (int kb_ = 0; kb_ < 4; ++kb_) {                               \
                v8h lo_ = *(const v8h*)(ab_ + kb_ * 32);                      \
                v8h hi_ = *(const v8h*)(ab_ + kb_ * 32 + 8);                  \
                _Pragma("unroll")                                             \
                for (int e_ = 0; e_ < 8; ++e_) {                              \
                    dst[kb_][e_] = lo_[e_];                                   \
                    dst[kb_][e_ + 8] = hi_[e_];                               \
                }                                                             \
            }                                                                 \
        }

    #define COMPUTE_STORE(src, j)                                             \
        {                                                                     \
            v8f acc_ = {};                                                    \
            _Pragma("unroll")                                                 \
            for (int kb_ = 0; kb_ < 4; ++kb_)                                 \
                acc_ = __builtin_amdgcn_wmma_f32_16x16x32_f16(                \
                    false, src[kb_], false, bf[kb_], (short)0, acc_,          \
                    false, false);                                            \
            if (ocol < QV) {                                                  \
                size_t rb_ = ((size_t)((mblk * 16 + (j)) * 16 + 8 * kgrp)) * QV + ocol; \
                _Pragma("unroll")                                             \
                for (int r_ = 0; r_ < 8; ++r_)                                \
                    out[rb_ + (size_t)r_ * QV] = acc_[r_];                    \
            }                                                                 \
        }

    v16h a0[4], a1[4];
    LOAD_A(a0, 0);
    __builtin_prefetch(abase0 + (size_t)2 * 16 * SCH, 0, 3);
    for (int jj = 0; jj < 16; jj += 2) {
        LOAD_A(a1, jj + 1);                 // overlap with compute of jj
        COMPUTE_STORE(a0, jj);
        if (jj + 2 < 16) {
            LOAD_A(a0, jj + 2);             // overlap with compute of jj+1
            __builtin_prefetch(abase0 + (size_t)(jj + 4) * 16 * SCH, 0, 3);
        }
        COMPUTE_STORE(a1, jj + 1);
    }
    #undef LOAD_A
    #undef COMPUTE_STORE
}

// ---------------- host-side orchestration ----------------
extern "C" void kernel_launch(void* const* d_in, const int* in_sizes, int n_in,
                              void* d_out, int out_size, void* d_ws, size_t ws_size,
                              hipStream_t stream) {
    const int*   idx    = (const int*)  d_in[0];
    const float* pre_W  = (const float*)d_in[1];
    const float* filt_W = (const float*)d_in[2];
    const float* gate_W = (const float*)d_in[3];
    const float* res_W  = (const float*)d_in[4];
    const float* skip_W = (const float*)d_in[5];
    const float* postW1 = (const float*)d_in[6];
    const float* postW2 = (const float*)d_in[7];
    float* out = (float*)d_out;

    // workspace layout (256B aligned)
    char* ws = (char*)d_ws;
    float*    x0   = (float*)   (ws + 0);          // 4*2047*32 f32 = 1,048,064 B
    float*    x1   = (float*)   (ws + 1048576);
    float*    skip = (float*)   (ws + 2097152);    // 4*1792*128 f32 = 3,670,016 B
    _Float16* Af16 = (_Float16*)(ws + 5767168);    // 7168*128 f16 = 1,835,008 B
    _Float16* Bp   = (_Float16*)(ws + 7602176);    // 1019*4*32*16 f16 = 4,173,824 B

    // 1) zero skip accumulator
    {
        int n = BB * OUTW * SCH;
        zero_kernel<<<(n + 255) / 256, 256, 0, stream>>>(skip, n);
    }
    // 2) embedding gather
    {
        int n = BB * TFULL * RCH;
        embed_kernel<<<(n + 255) / 256, 256, 0, stream>>>(idx, pre_W, x0);
    }
    // 3) pack post_W2 into WMMA B-fragment layout (independent of layers)
    {
        int n = NTILES * 4 * 32;
        packw2_kernel<<<(n + 255) / 256, 256, 0, stream>>>(postW2, Bp);
    }
    // 4) 8 sequential gated layers (ping-pong x buffers)
    float* xb[2] = { x0, x1 };
    for (int i = 0; i < NLAY; ++i) {
        int len_out  = (TFULL - 1) - i;   // 2046 - i
        int skip_off = len_out - OUTW;    // 254 - i
        int n = BB * len_out;
        layer_kernel<<<(n + 255) / 256, 256, 0, stream>>>(
            xb[i & 1], xb[(i + 1) & 1], skip,
            filt_W + (size_t)i * 2 * RCH * DCH,
            gate_W + (size_t)i * 2 * RCH * DCH,
            res_W  + (size_t)i * DCH * RCH,
            skip_W + (size_t)i * DCH * SCH,
            len_out, skip_off);
    }
    // 5) relu -> W1 -> relu, f16 conversion
    post_kernel<<<MROWS, 128, 0, stream>>>(skip, postW1, Af16);
    // 6) output-bandwidth-bound WMMA GEMM (467 MB of logits)
    gemm_kernel<<<dim3(NTILES, MTILES / 16), 32, 0, stream>>>(Af16, Bp, out);
}
